// GPUOptimizedAttention_87617332839004
// MI455X (gfx1250) — matrix operand order repaired and checked by reading, not compiled, not run on hardware
//
#include <hip/hip_runtime.h>
#include <math.h>

// ---------------------------------------------------------------------------
// MI455X (gfx1250) GQA attention block.
// B=2, S=2048, D=2048, H=32, KVH=8, HD=64, G=4, SCALE=1/8.
// Compute-bound (~155 GFLOP vs ~110MB HBM) -> all matmuls on
// v_wmma_f32_16x16x32_bf16 with f32 accumulation.
//  * weights pre-transposed to K-major (L2-resident) => GEMMs need no LDS
//  * V stored transposed by its projection GEMM => attention PV operand is
//    K-major in global memory
//  * flash softmax: cross-lane shuffle reductions, running m/l in registers
//  * K/V tiles staged to LDS with global_load_async_to_lds_b128 (ASYNCcnt)
//    when the builtin exists; synchronous b128 copies otherwise
// ---------------------------------------------------------------------------

typedef __bf16 bf16_t;
typedef __attribute__((ext_vector_type(16))) __bf16 v16bf;
typedef __attribute__((ext_vector_type(8)))  __bf16 bf16x8;
typedef __attribute__((ext_vector_type(8)))  float  v8f;
typedef __attribute__((ext_vector_type(4)))  int    v4i;

#define B_SZ    2
#define S_LEN   2048
#define D_MOD   2048
#define NHEAD   32
#define KVHEAD  8
#define HDIM    64
#define MROWS   4096           // B*S
#define LDQ     2048           // NHEAD*HDIM
#define LDKV    512            // KVHEAD*HDIM
#define SCALE_C 0.125f         // 1/sqrt(64)

#if __has_builtin(__builtin_amdgcn_global_load_async_to_lds_b128) && \
    __has_builtin(__builtin_amdgcn_s_wait_asynccnt)
#define HAVE_ASYNC_LDS 1
#else
#define HAVE_ASYNC_LDS 0
#endif

__device__ __forceinline__ v8f wmma_bf16(v16bf a, v16bf b, v8f c) {
  return __builtin_amdgcn_wmma_f32_16x16x32_bf16(
      false, a, false, b, (short)0, c, false, false);
}

// A fragment (16x32 MxK) from row-major source (element (m,k) at base[m*ld+k]).
// Lane L: row = L%16; lanes 0-15 hold K {0..7,16..23}, lanes 16-31 {8..15,24..31}
// => two contiguous 16B chunks per lane.
__device__ __forceinline__ v16bf load_A_rowmajor(const bf16_t* base, int ld, int lane) {
  const int half = lane >> 4, row = lane & 15;
  const bf16_t* p = base + (size_t)row * ld + half * 8;
  bf16x8 lo = *(const bf16x8*)p;
  bf16x8 hi = *(const bf16x8*)(p + 16);
  v16bf a;
#pragma unroll
  for (int j = 0; j < 8; ++j) { a[j] = lo[j]; a[8 + j] = hi[j]; }
  return a;
}

// B fragment (32x16 KxN) from K-major source (element (k,n) at base[n*ld+k]).
// Lane L: col = L%16; lanes 0-15 hold K {0..7,16..23}, lanes 16-31 {8..15,24..31}
__device__ __forceinline__ v16bf load_B_kmajor(const bf16_t* base, int ld, int lane) {
  const int col = lane & 15, koff = (lane >> 4) << 3;
  const bf16_t* p = base + (size_t)col * ld + koff;
  bf16x8 lo = *(const bf16x8*)p;
  bf16x8 hi = *(const bf16x8*)(p + 16);
  v16bf b;
#pragma unroll
  for (int j = 0; j < 8; ++j) { b[j] = lo[j]; b[8 + j] = hi[j]; }
  return b;
}

#if HAVE_ASYNC_LDS
// builtin signature (from hipcc diagnostic): (v4i AS1*, v4i AS3*, Iint, Iint)
typedef __attribute__((address_space(1))) v4i as1_v4i;
typedef __attribute__((address_space(3))) v4i as3_v4i;
__device__ __forceinline__ void async_copy16(const bf16_t* g, bf16_t* l) {
  __builtin_amdgcn_global_load_async_to_lds_b128(
      (as1_v4i*)g, (as3_v4i*)l, 0, 0);
}
#endif

// ---------------------------------------------------------------------------
// fp32 -> bf16 cast (grid-stride)
// ---------------------------------------------------------------------------
__global__ void cast_f32_bf16(const float* __restrict__ in,
                              bf16_t* __restrict__ out, long long n) {
  long long i = (long long)blockIdx.x * blockDim.x + threadIdx.x;
  long long stride = (long long)gridDim.x * blockDim.x;
  for (; i < n; i += stride) out[i] = (bf16_t)in[i];
}

// ---------------------------------------------------------------------------
// fp32 [K][N] -> bf16 [N][K] cast+transpose (32x32 LDS tile)
// ---------------------------------------------------------------------------
__global__ __launch_bounds__(256)
void cast_transpose_kernel(const float* __restrict__ in, bf16_t* __restrict__ out,
                           int K, int N) {
  __shared__ bf16_t tile[32][33];
  const int n0 = blockIdx.x * 32, k0 = blockIdx.y * 32;
  const int tx = threadIdx.x, ty = threadIdx.y;   // 32 x 8
#pragma unroll
  for (int j = 0; j < 32; j += 8)
    tile[ty + j][tx] = (bf16_t)in[(size_t)(k0 + ty + j) * N + n0 + tx];
  __syncthreads();
#pragma unroll
  for (int j = 0; j < 32; j += 8)
    out[(size_t)(n0 + ty + j) * K + k0 + tx] = tile[tx][ty + j];
}

// ---------------------------------------------------------------------------
// LDS-free WMMA GEMM: Y[M,N] = X[M,K] @ Wt[N,K]^T  (Wt is K-major, L2 resident)
// 4 waves, each wave owns a 32x64 strip (8 accumulators, 8 WMMA / 32-k-step).
// TRANS_OUT stores Y transposed as Y^T[N][M] (16B vector stores).
// ---------------------------------------------------------------------------
template <bool OUT_F32, bool TRANS_OUT>
__global__ __launch_bounds__(128)
void gemm_wt_kernel(const bf16_t* __restrict__ X, const bf16_t* __restrict__ Wt,
                    float* __restrict__ Yf, bf16_t* __restrict__ Yb,
                    int M, int N, int K) {
  const int tid = threadIdx.x, lane = tid & 31, wave = tid >> 5;
  const int n0 = blockIdx.x * 64;
  const int m0 = blockIdx.y * 128 + wave * 32;

  v8f acc[2][4] = {};
  for (int k0 = 0; k0 < K; k0 += 32) {
    v16bf a0 = load_A_rowmajor(X + (size_t)m0 * K + k0, K, lane);
    v16bf a1 = load_A_rowmajor(X + (size_t)(m0 + 16) * K + k0, K, lane);
#pragma unroll
    for (int c = 0; c < 4; ++c) {
      v16bf bq = load_B_kmajor(Wt + (size_t)(n0 + c * 16) * K + k0, K, lane);
      acc[0][c] = wmma_bf16(a0, bq, acc[0][c]);
      acc[1][c] = wmma_bf16(a1, bq, acc[1][c]);
    }
  }

  const int half = lane >> 4, col = lane & 15;
#pragma unroll
  for (int rb = 0; rb < 2; ++rb) {
#pragma unroll
    for (int c = 0; c < 4; ++c) {
      const int cn = n0 + c * 16 + col;
      const int rowb = m0 + rb * 16 + 8 * half;
      if (TRANS_OUT) {
        bf16x8 ov;
#pragma unroll
        for (int v = 0; v < 8; ++v) ov[v] = (bf16_t)acc[rb][c][v];
        *(bf16x8*)(Yb + (size_t)cn * M + rowb) = ov;
      } else {
#pragma unroll
        for (int v = 0; v < 8; ++v) {
          if (OUT_F32) Yf[(size_t)(rowb + v) * N + cn] = acc[rb][c][v];
          else         Yb[(size_t)(rowb + v) * N + cn] = (bf16_t)acc[rb][c][v];
        }
      }
    }
  }
}

// ---------------------------------------------------------------------------
// Flash attention: grid (S/64, H, B), 4 waves x 16 q-rows.
// Q [B*S, H*HD] bf16 ; K [B*S, KVH*HD] bf16 ; Vt [KVH*HD, B*S] bf16 (K-major).
// K/V tiles (shared by all 4 waves) staged to LDS asynchronously.
// ---------------------------------------------------------------------------
__global__ __launch_bounds__(128)
void attn_kernel(const bf16_t* __restrict__ Q, const bf16_t* __restrict__ Km,
                 const bf16_t* __restrict__ Vtg, bf16_t* __restrict__ O) {
  __shared__ bf16_t KL[64][64];        // [key][hd]  (K-major for QK^T)
  __shared__ bf16_t VtL[64][64];       // [hd][key]  (K-major for PV)
  __shared__ bf16_t Pt[4][16][64];     // per-wave prob tile

  const int tid = threadIdx.x, lane = tid & 31, wave = tid >> 5;
  const int h = blockIdx.y, b = blockIdx.z;
  const int kvh = h >> 2;              // G = 4
  const int half = lane >> 4, col = lane & 15;

  const size_t qrow0 = (size_t)b * S_LEN + blockIdx.x * 64 + wave * 16;
  const bf16_t* qbase = Q + qrow0 * LDQ + h * HDIM;
  v16bf qa0 = load_A_rowmajor(qbase, LDQ, lane);
  v16bf qa1 = load_A_rowmajor(qbase + 32, LDQ, lane);

  v8f o_acc[4] = {};
  float m_i[8], l_i[8];
#pragma unroll
  for (int v = 0; v < 8; ++v) { m_i[v] = -__builtin_inff(); l_i[v] = 0.f; }

  for (int kt = 0; kt < S_LEN; kt += 64) {
    const size_t krow0 = (size_t)b * S_LEN + kt;   // also V^T column base

    // prefetch next tile into cache hierarchy
    if (kt + 64 < S_LEN) {
      __builtin_prefetch(Km + (krow0 + 64 + (tid >> 1)) * LDKV + kvh * HDIM, 0, 1);
      __builtin_prefetch(Vtg + (size_t)(kvh * HDIM + (tid >> 1)) * MROWS + krow0 + 64, 0, 1);
    }

    // stage K tile [key][hd] and V^T tile [hd][key]: byte-identical copies
#pragma unroll
    for (int it = 0; it < 4; ++it) {
      const int idx = it * 128 + tid;   // 0..511
      const int row = idx >> 3;         // 0..63
      const int sc  = (idx & 7) * 8;    // 0,8,..,56
      const bf16_t* gk = Km  + (krow0 + row) * LDKV + kvh * HDIM + sc;
      const bf16_t* gv = Vtg + (size_t)(kvh * HDIM + row) * MROWS + krow0 + sc;
#if HAVE_ASYNC_LDS
      async_copy16(gk, &KL[row][sc]);
      async_copy16(gv, &VtL[row][sc]);
#else
      *(bf16x8*)&KL[row][sc]  = *(const bf16x8*)gk;
      *(bf16x8*)&VtL[row][sc] = *(const bf16x8*)gv;
#endif
    }
#if HAVE_ASYNC_LDS
    __builtin_amdgcn_s_wait_asynccnt(0);
#endif
    __syncthreads();

    // S = Q @ K^T   (scores in C/D layout: row = v + 8*half, col = key%16)
    v8f s[4] = {};
#pragma unroll
    for (int c = 0; c < 4; ++c) {
      v16bf kb0 = load_B_kmajor(&KL[c * 16][0], 64, lane);
      v16bf kb1 = load_B_kmajor(&KL[c * 16][32], 64, lane);
      s[c] = wmma_bf16(qa0, kb0, s[c]);
      s[c] = wmma_bf16(qa1, kb1, s[c]);
    }
#pragma unroll
    for (int c = 0; c < 4; ++c)
#pragma unroll
      for (int v = 0; v < 8; ++v) s[c][v] *= SCALE_C;

    // online softmax: per v-row, reduce across the 16-lane half via shuffles
#pragma unroll
    for (int v = 0; v < 8; ++v) {
      float t = fmaxf(fmaxf(s[0][v], s[1][v]), fmaxf(s[2][v], s[3][v]));
#pragma unroll
      for (int off = 8; off >= 1; off >>= 1) t = fmaxf(t, __shfl_xor(t, off, 16));
      const float mn = fmaxf(m_i[v], t);
      const float corr = __expf(m_i[v] - mn);
      m_i[v] = mn;
      float rs = 0.f;
#pragma unroll
      for (int c = 0; c < 4; ++c) {
        float p = __expf(s[c][v] - mn);
        s[c][v] = p;
        rs += p;
      }
#pragma unroll
      for (int off = 8; off >= 1; off >>= 1) rs += __shfl_xor(rs, off, 16);
      l_i[v] = l_i[v] * corr + rs;
#pragma unroll
      for (int c = 0; c < 4; ++c) o_acc[c][v] *= corr;
    }

    // P -> LDS (C/D order), reload as A fragments (cross-lane transpose)
#pragma unroll
    for (int c = 0; c < 4; ++c)
#pragma unroll
      for (int v = 0; v < 8; ++v)
        Pt[wave][v + 8 * half][c * 16 + col] = (bf16_t)s[c][v];

    v16bf pa0 = load_A_rowmajor(&Pt[wave][0][0], 64, lane);
    v16bf pa1 = load_A_rowmajor(&Pt[wave][0][32], 64, lane);

    // O += P @ V
#pragma unroll
    for (int c = 0; c < 4; ++c) {
      v16bf vb0 = load_B_kmajor(&VtL[c * 16][0], 64, lane);
      v16bf vb1 = load_B_kmajor(&VtL[c * 16][32], 64, lane);
      o_acc[c] = wmma_bf16(pa0, vb0, o_acc[c]);
      o_acc[c] = wmma_bf16(pa1, vb1, o_acc[c]);
    }
    __syncthreads();   // protect KL/VtL before next tile's staging
  }

  // normalize + store O (bf16)
  bf16_t* obase = O + qrow0 * LDQ + h * HDIM;
#pragma unroll
  for (int v = 0; v < 8; ++v) {
    const float linv = 1.f / l_i[v];
    const int r = v + 8 * half;
#pragma unroll
    for (int c = 0; c < 4; ++c)
      obase[(size_t)r * LDQ + c * 16 + col] = (bf16_t)(o_acc[c][v] * linv);
  }
}

// ---------------------------------------------------------------------------
// Host launcher
// ---------------------------------------------------------------------------
extern "C" void kernel_launch(void* const* d_in, const int* in_sizes, int n_in,
                              void* d_out, int out_size, void* d_ws, size_t ws_size,
                              hipStream_t stream) {
  const float* hs = (const float*)d_in[0];
  const float* Wq = (const float*)d_in[1];
  const float* Wk = (const float*)d_in[2];
  const float* Wv = (const float*)d_in[3];
  const float* Wo = (const float*)d_in[4];
  float* out = (float*)d_out;

  char* ws = (char*)d_ws;
  size_t off = 0;
  auto alloc = [&](size_t bytes) -> char* {
    char* p = ws + off;
    off += bytes;
    off = (off + 255) & ~(size_t)255;
    return p;
  };

  bf16_t* Xb  = (bf16_t*)alloc((size_t)MROWS * D_MOD * 2);
  bf16_t* WqT = (bf16_t*)alloc((size_t)2048 * D_MOD * 2);   // [N][K]
  bf16_t* WkT = (bf16_t*)alloc((size_t)512 * D_MOD * 2);
  bf16_t* WvT = (bf16_t*)alloc((size_t)512 * D_MOD * 2);
  bf16_t* WoT = (bf16_t*)alloc((size_t)D_MOD * 2048 * 2);
  bf16_t* Qb  = (bf16_t*)alloc((size_t)MROWS * 2048 * 2);
  bf16_t* Kb  = (bf16_t*)alloc((size_t)MROWS * 512 * 2);
  bf16_t* Vtg = (bf16_t*)alloc((size_t)512 * MROWS * 2);    // V^T [KVH*HD][B*S]
  bf16_t* Ob  = (bf16_t*)alloc((size_t)MROWS * 2048 * 2);

  // casts / weight transposes
  cast_f32_bf16<<<1024, 256, 0, stream>>>(hs, Xb, (long long)MROWS * D_MOD);
  dim3 tb(32, 8);
  cast_transpose_kernel<<<dim3(2048 / 32, D_MOD / 32), tb, 0, stream>>>(Wq, WqT, D_MOD, 2048);
  cast_transpose_kernel<<<dim3(512 / 32, D_MOD / 32), tb, 0, stream>>>(Wk, WkT, D_MOD, 512);
  cast_transpose_kernel<<<dim3(512 / 32, D_MOD / 32), tb, 0, stream>>>(Wv, WvT, D_MOD, 512);
  cast_transpose_kernel<<<dim3(D_MOD / 32, 2048 / 32), tb, 0, stream>>>(Wo, WoT, 2048, D_MOD);

  dim3 blk(128);
  // projections (V stored transposed)
  gemm_wt_kernel<false, false><<<dim3(2048 / 64, MROWS / 128), blk, 0, stream>>>(
      Xb, WqT, nullptr, Qb, MROWS, 2048, D_MOD);
  gemm_wt_kernel<false, false><<<dim3(512 / 64, MROWS / 128), blk, 0, stream>>>(
      Xb, WkT, nullptr, Kb, MROWS, 512, D_MOD);
  gemm_wt_kernel<false, true><<<dim3(512 / 64, MROWS / 128), blk, 0, stream>>>(
      Xb, WvT, nullptr, Vtg, MROWS, 512, D_MOD);

  // flash attention
  attn_kernel<<<dim3(S_LEN / 64, NHEAD, B_SZ), blk, 0, stream>>>(Qb, Kb, Vtg, Ob);

  // output projection (fp32 out)
  gemm_wt_kernel<true, false><<<dim3(D_MOD / 64, MROWS / 128), blk, 0, stream>>>(
      Ob, WoT, out, nullptr, MROWS, D_MOD, 2048);
}
